// CovalentLayer_23218593202238
// MI455X (gfx1250) — matrix operand order repaired and checked by reading, not compiled, not run on hardware
//
#include <hip/hip_runtime.h>
#include <hip/hip_bf16.h>

#define N_NODES 50000
#define E_EDGES 600000
#define E2_EDGES 600000
#define H 128
#define NH 8
#define DH 16

typedef __attribute__((ext_vector_type(16))) _Float16 v16h;
typedef __attribute__((ext_vector_type(8)))  _Float16 v8h;
typedef __attribute__((ext_vector_type(4)))  _Float16 v4h;
typedef __attribute__((ext_vector_type(8)))  float    v8f;

// ---------------- workspace layout (bytes) ----------------
static constexpr size_t WS_WN2E  = 0;                          // 128*384*2
static constexpr size_t WS_WC1   = WS_WN2E + 98304;            // 128*256*2
static constexpr size_t WS_WA    = WS_WC1  + 65536;            // 128*384*2
static constexpr size_t WS_WC2   = WS_WA   + 98304;            // 128*256*2
static constexpr size_t WS_BIASA = WS_WC2  + 65536;            // 512
static constexpr size_t WS_HE0   = WS_BIASA + 512;             // E*H*4
static constexpr size_t WS_AGG   = WS_HE0  + (size_t)E_EDGES*H*4;
static constexpr size_t WS_XJ    = WS_AGG  + (size_t)E_EDGES*H*4;
static constexpr size_t WS_BETA  = WS_XJ   + (size_t)E_EDGES*H*4;   // E*NH*4
static constexpr size_t WS_M     = WS_BETA + (size_t)E_EDGES*NH*4;  // N*NH*4
static constexpr size_t WS_DEN   = WS_M    + (size_t)N_NODES*NH*4;  // N*NH*4
static constexpr size_t WS_AGG2  = WS_DEN  + (size_t)N_NODES*NH*4;  // N*H*4

__device__ inline void atomAddF32(float* p, float v) {
  __hip_atomic_fetch_add(p, v, __ATOMIC_RELAXED, __HIP_MEMORY_SCOPE_AGENT);
}

// ---------------- prep: weights -> f16 transposed [n][k] ----------------
__global__ __launch_bounds__(256) void prep_weights(
    const float* __restrict__ Wn2e, const float* __restrict__ Wc1,
    const float* __restrict__ A1w,  const float* __restrict__ A2w,
    const float* __restrict__ A3w,  const float* __restrict__ Wc2,
    const float* __restrict__ A1b,  const float* __restrict__ A2b,
    const float* __restrict__ A3b,
    _Float16* wt_n2e, _Float16* wt_c1, _Float16* wt_A, _Float16* wt_c2,
    float* biasA) {
  int idx = blockIdx.x * 256 + threadIdx.x;
  if (idx < 128 * 384) {
    int n = idx / 384, k = idx % 384;
    wt_n2e[idx] = (_Float16)Wn2e[k * 128 + n];
    float av = (k < 128) ? A1w[k * 128 + n]
             : (k < 256) ? A2w[(k - 128) * 128 + n]
                         : A3w[(k - 256) * 128 + n];
    wt_A[idx] = (_Float16)av;
  }
  if (idx < 128 * 256) {
    int n = idx / 256, k = idx % 256;
    wt_c1[idx] = (_Float16)Wc1[k * 128 + n];
    wt_c2[idx] = (_Float16)Wc2[k * 128 + n];
  }
  if (idx < 128) biasA[idx] = A1b[idx] + A2b[idx] + A3b[idx];
}

// ---------------- zero / -inf init ----------------
__global__ __launch_bounds__(256) void init_buffers(float* agg, float* agg2,
                                                    float* den, float* m) {
  long long i = (long long)blockIdx.x * 256 + threadIdx.x;
  const long long n1 = (long long)E_EDGES * H / 4;
  const long long n2 = (long long)N_NODES * H / 4;
  const long long n3 = (long long)N_NODES * NH / 4;
  float4 z = make_float4(0.f, 0.f, 0.f, 0.f);
  if (i < n1) { ((float4*)agg)[i] = z; return; }
  i -= n1;
  if (i < n2) { ((float4*)agg2)[i] = z; return; }
  i -= n2;
  if (i < n3) { ((float4*)den)[i] = z; return; }
  i -= n3;
  if (i < n3) {
    float ninf = -__builtin_huge_valf();
    ((float4*)m)[i] = make_float4(ninf, ninf, ninf, ninf);
  }
}

// ---------------- fused WMMA GEMM ----------------
// A-tile: 64 rows x KH halves in LDS (each row = concat of 128-wide segments).
// 8 waves: rowTile = w&3 (16 rows), colGroup = w>>2 (64 cols = 4 WMMA col tiles).
template <int KH, bool GATHER, bool ADDSRC, bool GUARD>
__global__ __launch_bounds__(256) void fused_gemm(
    const float* __restrict__ srcA, const float* __restrict__ srcB,
    const float* __restrict__ srcC, const int* __restrict__ gidx0,
    const int* __restrict__ gidx1, const _Float16* __restrict__ WT,
    const float* __restrict__ bias, const float* __restrict__ addsrc,
    float* __restrict__ out, int nrows) {
  constexpr int NSEG = KH / 128;
  constexpr int STRIDE = KH + 8;  // halves; keeps 16B alignment, skews banks
  __shared__ _Float16 At[64 * STRIDE];
  __shared__ int sIdx0[64];
  __shared__ int sIdx1[64];

  const int tid = threadIdx.x;
  const long long r0 = (long long)blockIdx.x * 64;

  if constexpr (GATHER) {
    if (tid < 64) {
      long long e = r0 + tid;
      sIdx0[tid] = (!GUARD || e < nrows) ? gidx0[e] : 0;
    } else if (tid < 128) {
      int s = tid - 64;
      long long e = r0 + s;
      sIdx1[s] = (!GUARD || e < nrows) ? gidx1[e] : 0;
    }
    __syncthreads();
  }

  // stage A tile (fp32 -> fp16)
  const int TOT4 = 64 * NSEG * 32;
  for (int idx = tid; idx < TOT4; idx += 256) {
    int s = idx / (NSEG * 32);
    int rr = idx % (NSEG * 32);
    int seg = rr >> 5, c4 = rr & 31;
    long long e = r0 + s;
    float4 f = make_float4(0.f, 0.f, 0.f, 0.f);
    if (!GUARD || e < nrows) {
      const float* p;
      if constexpr (GATHER) {
        if (seg == 0)      p = srcA + (long long)sIdx0[s] * H;
        else if (seg == 1) p = srcB + (long long)sIdx1[s] * H;
        else               p = srcC + e * H;
      } else {
        p = (seg == 0 ? srcA : srcB) + e * H;
      }
      f = ((const float4*)p)[c4];
    }
    v4h hv = {(_Float16)f.x, (_Float16)f.y, (_Float16)f.z, (_Float16)f.w};
    *(v4h*)(&At[s * STRIDE + seg * 128 + c4 * 4]) = hv;
  }
  __syncthreads();

  const int lane = tid & 31;
  const int w = tid >> 5;
  const int rowTile = w & 3;
  const int colGroup = w >> 2;
  const int m = rowTile * 16 + (lane & 15);
  const int hi = lane >> 4;

  v8f acc[4];
#pragma unroll
  for (int ct = 0; ct < 4; ct++) acc[ct] = (v8f){0, 0, 0, 0, 0, 0, 0, 0};

#pragma unroll
  for (int k0 = 0; k0 < KH; k0 += 32) {
    union { v16h v; v8h h[2]; } a;
    a.h[0] = *(const v8h*)(&At[m * STRIDE + k0 + hi * 16]);
    a.h[1] = *(const v8h*)(&At[m * STRIDE + k0 + hi * 16 + 8]);
#pragma unroll
    for (int ct = 0; ct < 4; ct++) {
      int n = colGroup * 64 + ct * 16 + (lane & 15);
      v16h b = *(const v16h*)(WT + (size_t)n * KH + k0 + hi * 16);
      acc[ct] = __builtin_amdgcn_wmma_f32_16x16x32_f16(
          false, a.v, false, b, (short)0, acc[ct], false, false);
    }
  }

  // epilogue: bias + ReLU (+ residual)
#pragma unroll
  for (int ct = 0; ct < 4; ct++) {
    int n = colGroup * 64 + ct * 16 + (lane & 15);
    float bi = bias[n];
#pragma unroll
    for (int r = 0; r < 8; r++) {
      long long gr = r0 + rowTile * 16 + hi * 8 + r;
      if (GUARD && gr >= nrows) continue;
      float v = acc[ct][r] + bi;
      v = v > 0.f ? v : 0.f;
      if constexpr (ADDSRC) v += addsrc[gr * H + n];
      out[gr * H + n] = v;
    }
  }
}

// ---------------- line-graph aggregation: agg[btgt] += he0[bsrc]*battr ------
__global__ __launch_bounds__(256) void line_agg(const float* __restrict__ he0,
                                                const float* __restrict__ battr,
                                                const int* __restrict__ bidx,
                                                float* agg) {
  long long idx = (long long)blockIdx.x * 256 + threadIdx.x;
  if (idx >= (long long)E2_EDGES * 32) return;
  long long e = idx >> 5;
  int c4 = idx & 31;
  int s = bidx[e];
  int t = bidx[E2_EDGES + e];
  float4 hv = ((const float4*)(he0 + (long long)s * H))[c4];
  float4 bv = ((const float4*)(battr + e * H))[c4];
  float* dst = agg + (long long)t * H + c4 * 4;
  atomAddF32(dst + 0, hv.x * bv.x);
  atomAddF32(dst + 1, hv.y * bv.y);
  atomAddF32(dst + 2, hv.z * bv.z);
  atomAddF32(dst + 3, hv.w * bv.w);
}

// ---------------- attention: xj = x[src]*he ; beta = leaky(attn) ------------
__global__ __launch_bounds__(256) void attn_beta(
    const float* __restrict__ x, const int* __restrict__ eidx,
    const float* __restrict__ he, const float* __restrict__ attw,
    const float* __restrict__ attb, float* __restrict__ xj,
    float* __restrict__ beta) {
  long long idx = (long long)blockIdx.x * 256 + threadIdx.x;
  if (idx >= (long long)E_EDGES * NH) return;
  long long e = idx >> 3;
  int h = idx & 7;
  int s = eidx[e];
  int t = eidx[E_EDGES + e];
  const float4* xs = (const float4*)(x + (long long)s * H + h * DH);
  const float4* xt = (const float4*)(x + (long long)t * H + h * DH);
  const float4* hp = (const float4*)(he + e * H + h * DH);
  float4* xjp = (float4*)(xj + e * H + h * DH);
  float acc = attb[0];
#pragma unroll
  for (int q = 0; q < 4; q++) {
    float4 a = xs[q], hh = hp[q], b = xt[q];
    float4 j = make_float4(a.x * hh.x, a.y * hh.y, a.z * hh.z, a.w * hh.w);
    xjp[q] = j;
    float4 w1 = ((const float4*)attw)[q];
    float4 w2 = ((const float4*)attw)[4 + q];
    acc += j.x * w1.x + j.y * w1.y + j.z * w1.z + j.w * w1.w;
    acc += b.x * w2.x + b.y * w2.y + b.z * w2.z + b.w * w2.w;
  }
  beta[idx] = acc > 0.f ? acc : 0.01f * acc;
}

// ---------------- segment max (float atomic max via int encoding) -----------
__global__ __launch_bounds__(256) void seg_max(const int* __restrict__ eidx,
                                               const float* __restrict__ beta,
                                               float* m) {
  long long idx = (long long)blockIdx.x * 256 + threadIdx.x;
  if (idx >= (long long)E_EDGES * NH) return;
  long long e = idx >> 3;
  int h = idx & 7;
  int t = eidx[E_EDGES + e];
  float v = beta[idx];
  float* p = m + (long long)t * NH + h;
  if (v >= 0.f) atomicMax((int*)p, __float_as_int(v));
  else          atomicMin((unsigned int*)p, __float_as_uint(v));
}

// ---------------- exp + denominator ----------------
__global__ __launch_bounds__(256) void seg_exp(const int* __restrict__ eidx,
                                               float* beta,  // in: beta, out: ex
                                               const float* __restrict__ m,
                                               float* den) {
  long long idx = (long long)blockIdx.x * 256 + threadIdx.x;
  if (idx >= (long long)E_EDGES * NH) return;
  long long e = idx >> 3;
  int h = idx & 7;
  int t = eidx[E_EDGES + e];
  float mm = m[(long long)t * NH + h];
  if (!__builtin_isfinite(mm)) mm = 0.f;
  float ex = __expf(beta[idx] - mm);
  beta[idx] = ex;
  atomAddF32(den + (long long)t * NH + h, ex);
}

// ---------------- alpha * xj scatter-add ----------------
__global__ __launch_bounds__(256) void seg_scatter(
    const int* __restrict__ eidx, const float* __restrict__ ex,
    const float* __restrict__ den, const float* __restrict__ xj, float* agg2) {
  long long idx = (long long)blockIdx.x * 256 + threadIdx.x;
  if (idx >= (long long)E_EDGES * NH) return;
  long long e = idx >> 3;
  int h = idx & 7;
  int t = eidx[E_EDGES + e];
  float alpha = ex[idx] / (den[(long long)t * NH + h] + 1e-16f);
  const float4* xp = (const float4*)(xj + e * H + h * DH);
  float* dst = agg2 + (long long)t * H + h * DH;
#pragma unroll
  for (int q = 0; q < 4; q++) {
    float4 v = xp[q];
    atomAddF32(dst + q * 4 + 0, alpha * v.x);
    atomAddF32(dst + q * 4 + 1, alpha * v.y);
    atomAddF32(dst + q * 4 + 2, alpha * v.z);
    atomAddF32(dst + q * 4 + 3, alpha * v.w);
  }
}

extern "C" void kernel_launch(void* const* d_in, const int* in_sizes, int n_in,
                              void* d_out, int out_size, void* d_ws,
                              size_t ws_size, hipStream_t stream) {
  (void)in_sizes; (void)n_in; (void)out_size; (void)ws_size;
  const float* x     = (const float*)d_in[0];
  const int*   eidx  = (const int*)d_in[1];
  const float* eattr = (const float*)d_in[2];
  const int*   bidx  = (const int*)d_in[3];
  const float* battr = (const float*)d_in[4];
  const float* Wn2e  = (const float*)d_in[5];
  const float* bn2e  = (const float*)d_in[6];
  const float* A1w   = (const float*)d_in[7];
  const float* A1b   = (const float*)d_in[8];
  const float* A2w   = (const float*)d_in[9];
  const float* A2b   = (const float*)d_in[10];
  const float* A3w   = (const float*)d_in[11];
  const float* A3b   = (const float*)d_in[12];
  const float* Wc1w  = (const float*)d_in[13];
  const float* Wc1b  = (const float*)d_in[14];
  const float* attw  = (const float*)d_in[15];
  const float* attb  = (const float*)d_in[16];
  const float* Wc2w  = (const float*)d_in[17];
  const float* Wc2b  = (const float*)d_in[18];

  char* ws = (char*)d_ws;
  _Float16* wt_n2e = (_Float16*)(ws + WS_WN2E);
  _Float16* wt_c1  = (_Float16*)(ws + WS_WC1);
  _Float16* wt_A   = (_Float16*)(ws + WS_WA);
  _Float16* wt_c2  = (_Float16*)(ws + WS_WC2);
  float* biasA = (float*)(ws + WS_BIASA);
  float* he0   = (float*)(ws + WS_HE0);
  float* agg   = (float*)(ws + WS_AGG);
  float* xj    = (float*)(ws + WS_XJ);
  float* beta  = (float*)(ws + WS_BETA);
  float* m     = (float*)(ws + WS_M);
  float* den   = (float*)(ws + WS_DEN);
  float* agg2  = (float*)(ws + WS_AGG2);

  float* out_hx = (float*)d_out;
  float* out_he = out_hx + (size_t)N_NODES * H;
  float* out_ha = out_he + (size_t)E_EDGES * H;

  // 1) weights -> f16 transposed; fused A-bias
  prep_weights<<<192, 256, 0, stream>>>(Wn2e, Wc1w, A1w, A2w, A3w, Wc2w, A1b,
                                        A2b, A3b, wt_n2e, wt_c1, wt_A, wt_c2,
                                        biasA);
  // 2) zero agg/agg2/den, m = -inf
  long long initN = (long long)E_EDGES * H / 4 + (long long)N_NODES * H / 4 +
                    2 * ((long long)N_NODES * NH / 4);
  init_buffers<<<(unsigned)((initN + 255) / 256), 256, 0, stream>>>(agg, agg2,
                                                                    den, m);
  // 3) he0 = ReLU([x[src]|x[tgt]|eattr] @ Wn2e + b)
  fused_gemm<384, true, false, false><<<E_EDGES / 64, 256, 0, stream>>>(
      x, x, eattr, eidx, eidx + E_EDGES, wt_n2e, bn2e, nullptr, he0, E_EDGES);
  // 4) agg[btgt] += he0[bsrc] * battr
  line_agg<<<(unsigned)((long long)E2_EDGES * 32 / 256), 256, 0, stream>>>(
      he0, battr, bidx, agg);
  // 5) he = ReLU([he0|agg] @ Wc1 + b)   -> output
  fused_gemm<256, false, false, false><<<E_EDGES / 64, 256, 0, stream>>>(
      he0, agg, nullptr, nullptr, nullptr, wt_c1, Wc1b, nullptr, out_he,
      E_EDGES);
  // 6) ha = battr + ReLU([he0[bsrc]|he0[btgt]|battr] @ [A1;A2;A3] + biasA)
  fused_gemm<384, true, true, false><<<E2_EDGES / 64, 256, 0, stream>>>(
      he0, he0, battr, bidx, bidx + E2_EDGES, wt_A, biasA, battr, out_ha,
      E2_EDGES);
  // 7) attention logits
  attn_beta<<<E_EDGES * NH / 256, 256, 0, stream>>>(x, eidx, out_he, attw,
                                                    attb, xj, beta);
  // 8-10) segment softmax + weighted scatter
  seg_max<<<E_EDGES * NH / 256, 256, 0, stream>>>(eidx, beta, m);
  seg_exp<<<E_EDGES * NH / 256, 256, 0, stream>>>(eidx, beta, m, den);
  seg_scatter<<<E_EDGES * NH / 256, 256, 0, stream>>>(eidx, beta, den, xj,
                                                      agg2);
  // 11) hx = ReLU([x|agg2] @ Wc2 + b)
  fused_gemm<256, false, false, true><<<(N_NODES + 63) / 64, 256, 0, stream>>>(
      x, agg2, nullptr, nullptr, nullptr, wt_c2, Wc2b, nullptr, out_hx,
      N_NODES);
}
